// DeformConv2d_5961414607249
// MI455X (gfx1250) — matrix operand order repaired and checked by reading, not compiled, not run on hardware
//
#include <hip/hip_runtime.h>
#include <math.h>

// Problem constants (match reference)
constexpr int Bn   = 8;
constexpr int CIN  = 256;
constexpr int COUT = 256;
constexpr int H    = 56, W = 56, OH = 56, OW = 56;
constexpr int KT   = 9;                 // 3x3 taps
constexpr int HW   = H * W;
constexpr int OHW  = OH * OW;
constexpr int NPOS = Bn * OHW;          // 25088 = 196 * 128

// Tiling
constexpr int MTILE = 128;              // Cout tile
constexpr int NTILE = 128;              // position tile
constexpr int KC    = 32;               // K chunk (WMMA depth)
constexpr int NCH   = CIN / KC;         // 8 channel-chunks per tap
constexpr int NFRAG = NTILE / 16;       // 8 B-fragment blocks per N tile
constexpr int FGTOT = COUT / 16;        // 16 A-fragment row-blocks total

constexpr size_t WRF_ELEMS = (size_t)KT * NCH * FGTOT * 32 * 16;  // 589,824
constexpr size_t WRF_BYTES = WRF_ELEMS * 2;                       // 1.18 MB
constexpr size_t XT_ELEMS  = (size_t)Bn * HW * CIN;               // 6,422,528

typedef __attribute__((ext_vector_type(16))) _Float16 v16h;
typedef __attribute__((ext_vector_type(8)))  float    v8f;

// Pre-kernel 1: weight[oc][c][tap] (f32) -> fragment-ready f16 layout:
//   wrF[(((t*NCH + cb)*FGTOT + fg)*32 + lane)*16 + i]
// holding A[oc = fg*16 + (lane&15)][k = kperm(lane>>4, i)] in
// this code's A k order: k = half*16+i (contiguous, not AMD's table order; matches the B gather).
__global__ void build_wfrag(const float* __restrict__ w,
                            _Float16* __restrict__ wrF) {
  int idx = blockIdx.x * 256 + threadIdx.x;
  if (idx >= (int)WRF_ELEMS) return;
  const int i    = idx & 15;
  const int lane = (idx >> 4) & 31;
  const int fg   = (idx >> 9) & 15;
  const int cb   = (idx >> 13) & 7;
  const int t    = idx >> 16;
  const int half = lane >> 4, lm = lane & 15;
  const int oc = fg * 16 + lm;
  const int kk = half * 16 + i;
  const int c  = cb * KC + kk;
  wrF[idx] = (_Float16)w[(oc * CIN + c) * KT + t];
}

// Pre-kernel 2: x NCHW f32 -> xT NHWC f16 so channel gathers are contiguous.
//   xT[(b*HW + o)*CIN + c] = x[(b*CIN + c)*HW + o]
__global__ void build_xT(const float* __restrict__ x,
                         _Float16* __restrict__ xT) {
  int idx = blockIdx.x * 256 + threadIdx.x;
  if (idx >= (int)XT_ELEMS) return;
  const int c  = idx & (CIN - 1);
  const int bo = idx >> 8;                  // b*HW + o
  const int b  = bo / HW;
  const int o  = bo - b * HW;
  xT[idx] = (_Float16)x[((size_t)(b * CIN + c)) * HW + o];
}

__launch_bounds__(256)
__global__ void deform_conv2d_wmma(const _Float16* __restrict__ xT,
                                   const float* __restrict__ offset,
                                   const float* __restrict__ mask,
                                   const _Float16* __restrict__ wrF,
                                   const float* __restrict__ bias,
                                   float* __restrict__ out) {
  __shared__ int      s_xbase[NTILE];            // b * HW * CIN per position
  __shared__ int      s_oidx[KT][NTILE][4];      // clamped corner idx * CIN
  __shared__ _Float16 s_wgt [KT][NTILE][4];      // f16 bilinear wgt * mask
  __shared__ __align__(32) _Float16 s_colF[2][NFRAG][32][16];  // fragment-ready cols

  const int tid = threadIdx.x;
  const int n0  = blockIdx.x * NTILE;
  const int m0  = blockIdx.y * MTILE;

  // ---------------- Phase 0: sampling metadata for 128 pos x 9 taps ----------
  for (int idx = tid; idx < KT * NTILE; idx += 256) {
    const int t = idx / NTILE;
    const int p = idx - t * NTILE;
    const int n = n0 + p;
    const int b  = n / OHW;
    const int r  = n - b * OHW;
    const int oh = r / OW;
    const int ow = r - oh * OW;
    if (t == 0) s_xbase[p] = b * HW * CIN;

    const float offi = offset[(((b * (2 * KT)) + 2 * t    ) * OH + oh) * OW + ow];
    const float offj = offset[(((b * (2 * KT)) + 2 * t + 1) * OH + oh) * OW + ow];
    const float mval = mask  [(((b * KT)       + t        ) * OH + oh) * OW + ow];

    const float ci = offi + (float)(oh - 1 + t / 3);
    const float cj = offj + (float)(ow - 1 + t % 3);
    const float fli = floorf(ci), flj = floorf(cj);
    const float fi = ci - fli,  fj = cj - flj;
    const int i0 = (int)fli, j0 = (int)flj;
    const int i1 = i0 + 1,   j1 = j0 + 1;

    const bool vi0 = (i0 >= 0) & (i0 < H), vi1 = (i1 >= 0) & (i1 < H);
    const bool vj0 = (j0 >= 0) & (j0 < W), vj1 = (j1 >= 0) & (j1 < W);
    const int ci0 = min(max(i0, 0), H - 1), ci1 = min(max(i1, 0), H - 1);
    const int cj0 = min(max(j0, 0), W - 1), cj1 = min(max(j1, 0), W - 1);

    s_oidx[t][p][0] = (ci0 * W + cj0) * CIN;   // lt
    s_oidx[t][p][1] = (ci0 * W + cj1) * CIN;   // rt
    s_oidx[t][p][2] = (ci1 * W + cj0) * CIN;   // lb
    s_oidx[t][p][3] = (ci1 * W + cj1) * CIN;   // rb
    s_wgt[t][p][0] = (_Float16)((1.f - fi) * (1.f - fj) * mval * (float)(vi0 & vj0));
    s_wgt[t][p][1] = (_Float16)((1.f - fi) * fj         * mval * (float)(vi0 & vj1));
    s_wgt[t][p][2] = (_Float16)(fi         * (1.f - fj) * mval * (float)(vi1 & vj0));
    s_wgt[t][p][3] = (_Float16)(fi         * fj         * mval * (float)(vi1 & vj1));
  }
  __syncthreads();

  // ---------------- Phase 1: K loop with WMMA --------------------------------
  const int lane  = tid & 31;
  const int wave  = tid >> 5;
  const int waveM = wave >> 2;   // 0..1 -> 64 rows each
  const int waveN = wave & 3;    // 0..3 -> 32 cols each
  const int half  = lane >> 4;
  const int lm    = lane & 15;

  // Col-writer mapping: this thread produces lane `lane` of B-fragment block
  // `wave`: value(i) = col[k = half*16 + i][n = wave*16 + lm] -> contiguous 32B.
  const int pW = wave * 16 + lm;
  const int xb = s_xbase[pW];

  v8f acc[4][2];
  {
    v8f zero = {};
#pragma unroll
    for (int mb = 0; mb < 4; ++mb)
#pragma unroll
      for (int nb = 0; nb < 2; ++nb) acc[mb][nb] = zero;
  }

  const int fgBase = (m0 >> 4) + waveM * 4;   // this wave's first A row-block

  int buf = 0;
  for (int t = 0; t < KT; ++t) {
    const int o0 = s_oidx[t][pW][0], o1 = s_oidx[t][pW][1];
    const int o2 = s_oidx[t][pW][2], o3 = s_oidx[t][pW][3];
    const _Float16 h0 = s_wgt[t][pW][0], h1 = s_wgt[t][pW][1];
    const _Float16 h2 = s_wgt[t][pW][2], h3 = s_wgt[t][pW][3];

    for (int cb = 0; cb < NCH; ++cb) {
      const int c0 = cb * KC;

      // ---- gather 4 corners x 16 contiguous channels; blend in packed f16 ---
      const _Float16* xc = xT + xb + c0 + half * 16;
      const v16h g0 = *(const v16h*)(xc + o0);
      const v16h g1 = *(const v16h*)(xc + o1);
      const v16h g2 = *(const v16h*)(xc + o2);
      const v16h g3 = *(const v16h*)(xc + o3);
      const v16h cv = g0 * h0 + g1 * h1 + g2 * h2 + g3 * h3;

      // ---- A fragments straight from global (L2-resident, coalesced b128) ---
      const _Float16* aB =
          wrF + ((size_t)((t * NCH + cb) * FGTOT + fgBase) * 32 + lane) * 16;
      v16h afr[4];
#pragma unroll
      for (int mb = 0; mb < 4; ++mb)
        afr[mb] = *(const v16h*)(aB + (size_t)mb * 32 * 16);

      *(v16h*)(&s_colF[buf][wave][lane][0]) = cv;
      __syncthreads();

      // ---- B fragments: one contiguous 32B LDS read each ----
      v16h bfr[2];
#pragma unroll
      for (int nb = 0; nb < 2; ++nb)
        bfr[nb] = *(const v16h*)(&s_colF[buf][waveN * 2 + nb][lane][0]);

#pragma unroll
      for (int mb = 0; mb < 4; ++mb)
#pragma unroll
        for (int nb = 0; nb < 2; ++nb)
          acc[mb][nb] = __builtin_amdgcn_wmma_f32_16x16x32_f16(
              false, afr[mb], false, bfr[nb], (short)0, acc[mb][nb], false, false);

      buf ^= 1;
    }
  }

  // ---------------- Epilogue: bias + store [b, oc, oh, ow] -------------------
#pragma unroll
  for (int nb = 0; nb < 2; ++nb) {
    const int n = n0 + waveN * 32 + nb * 16 + lm;
    const int b = n / OHW;
    const int r = n - b * OHW;                 // oh*OW + ow
    float* obase = out + (size_t)b * COUT * OHW + r;
#pragma unroll
    for (int mb = 0; mb < 4; ++mb) {
      const int ocb = m0 + waveM * 64 + mb * 16 + half * 8;
#pragma unroll
      for (int rr = 0; rr < 8; ++rr) {
        const int oc = ocb + rr;
        obase[(size_t)oc * OHW] = acc[mb][nb][rr] + bias[oc];
      }
    }
  }
}

extern "C" void kernel_launch(void* const* d_in, const int* in_sizes, int n_in,
                              void* d_out, int out_size, void* d_ws, size_t ws_size,
                              hipStream_t stream) {
  const float* x      = (const float*)d_in[0];
  const float* offset = (const float*)d_in[1];
  const float* mask   = (const float*)d_in[2];
  const float* weight = (const float*)d_in[3];
  const float* bias   = (const float*)d_in[4];
  float*       out    = (float*)d_out;

  _Float16* wrF = (_Float16*)d_ws;                          // 1.18 MB
  _Float16* xT  = (_Float16*)((char*)d_ws + WRF_BYTES);     // 12.8 MB

  build_wfrag<<<((int)WRF_ELEMS + 255) / 256, 256, 0, stream>>>(weight, wrF);
  build_xT  <<<((int)XT_ELEMS  + 255) / 256, 256, 0, stream>>>(x, xT);

  dim3 grid(NPOS / NTILE /*196*/, COUT / MTILE /*2*/);
  deform_conv2d_wmma<<<grid, 256, 0, stream>>>(xT, offset, mask, wrF, bias, out);
}
